// GATEncoder_2259152797883
// MI455X (gfx1250) — matrix operand order repaired and checked by reading, not compiled, not run on hardware
//
#include <hip/hip_runtime.h>
#include <cmath>

// ---------------- problem constants (match reference) ----------------
#define NN    30000            // nodes
#define EE    480000           // edges before self loops
#define ETOT  (EE + NN)        // edges incl. self loops
#define FDIM  128              // feature dim for both GEMMs (IN=H*HID=OUT=128)
#define NEG_SLOPE 0.2f
#define SM_EPS    1e-16f

typedef __attribute__((ext_vector_type(16))) _Float16 v16h;
typedef __attribute__((ext_vector_type(8)))  _Float16 v8h;
typedef __attribute__((ext_vector_type(8)))  float    v8f;

// ---------------- helpers ----------------
static __device__ __forceinline__ void edge_sd(const int* __restrict__ ei, int e, int& s, int& d) {
  if (e < EE) { s = ei[e]; d = ei[EE + e]; }
  else        { s = e - EE; d = e - EE; }          // self loops appended
}

// float atomic max via monotonic int/uint bit trick (works for mixed signs,
// initial value -inf). Lowers to native global_atomic_max/min.
static __device__ __forceinline__ void atomic_max_f32(float* addr, float val) {
  if (val >= 0.0f) atomicMax((int*)addr, __float_as_int(val));
  else             atomicMin((unsigned int*)addr, __float_as_uint(val));
}

// ---------------- precision-split conversion ----------------
__global__ void split_kernel(const float* __restrict__ src, _Float16* __restrict__ hi,
                             _Float16* __restrict__ lo, int n) {
  int i = blockIdx.x * blockDim.x + threadIdx.x;
  if (i >= n) return;
  float v = src[i];
  _Float16 h = (_Float16)v;
  hi[i] = h;
  lo[i] = (_Float16)(v - (float)h);
}

// split + transpose (B matrix stored [K,N] row-major -> [N,K] so the GEMM's
// per-lane B gather becomes a contiguous 32-byte vector load)
__global__ void splitT_kernel(const float* __restrict__ src, _Float16* __restrict__ hiT,
                              _Float16* __restrict__ loT, int rows, int cols) {
  int i = blockIdx.x * blockDim.x + threadIdx.x;
  if (i >= rows * cols) return;
  int r = i / cols, c = i - r * cols;
  float v = src[i];
  _Float16 h = (_Float16)v;
  hiT[(size_t)c * rows + r] = h;
  loT[(size_t)c * rows + r] = (_Float16)(v - (float)h);
}

__global__ void fill_kernel(float* __restrict__ p, float v, int n) {
  int i = blockIdx.x * blockDim.x + threadIdx.x;
  if (i < n) p[i] = v;
}

// ---------------- WMMA GEMM: C[M,Ncol] = A[M,K] @ B[K,Ncol] ----------------
// A split row-major (hi+lo f16), B split AND transposed to [Ncol,K].
// One wave per 16x16 output tile; K-slabs of 32 with 3 WMMAs (hi*hi+hi*lo+lo*hi)
// for ~fp32 accuracy. Guard is wave-uniform so EXEC is all-ones at each WMMA.
__global__ void wmma_gemm_kernel(const _Float16* __restrict__ Ahi, const _Float16* __restrict__ Alo,
                                 const _Float16* __restrict__ BhiT, const _Float16* __restrict__ BloT,
                                 float* __restrict__ C, int M, int K, int Ncol) {
  int wave = (int)((blockIdx.x * blockDim.x + threadIdx.x) >> 5);
  int lane = threadIdx.x & 31;
  int ntiles = Ncol >> 4;
  int mt = wave / ntiles;
  int nt = wave - mt * ntiles;
  if (mt * 16 >= M) return;                         // uniform across the wave

  int mrow = mt * 16 + (lane & 15);
  int ncol = nt * 16 + (lane & 15);
  int grp  = lane >> 4;                             // 0: lanes 0-15, 1: lanes 16-31

  const _Float16* arh = Ahi  + (size_t)mrow * K;
  const _Float16* arl = Alo  + (size_t)mrow * K;
  const _Float16* brh = BhiT + (size_t)ncol * K;
  const _Float16* brl = BloT + (size_t)ncol * K;

  v8f acc = {};
  for (int kc = 0; kc < K; kc += 32) {
    // A 16x32 f16 layout: lane grp g, element j -> K = kc + 16g + j (contiguous, matches B)
    int ka0 = kc + grp * 16;
    v8h a0h = *(const v8h*)(arh + ka0);
    v8h a1h = *(const v8h*)(arh + ka0 + 8);
    v8h a0l = *(const v8h*)(arl + ka0);
    v8h a1l = *(const v8h*)(arl + ka0 + 8);
    v16h ah, al;
#pragma unroll
    for (int j = 0; j < 8; ++j) {
      ah[j] = a0h[j]; ah[j + 8] = a1h[j];
      al[j] = a0l[j]; al[j + 8] = a1l[j];
    }
    // B 32x16 f16 layout: lane grp g, element j -> K = kc + 16g + j (contiguous)
    int kb0 = kc + grp * 16;
    v16h bh = *(const v16h*)(brh + kb0);
    v16h bl = *(const v16h*)(brl + kb0);

    acc = __builtin_amdgcn_wmma_f32_16x16x32_f16(false, ah, false, bh, (short)0, acc, false, false);
    acc = __builtin_amdgcn_wmma_f32_16x16x32_f16(false, ah, false, bl, (short)0, acc, false, false);
    acc = __builtin_amdgcn_wmma_f32_16x16x32_f16(false, al, false, bh, (short)0, acc, false, false);
  }
  // C/D 16x16 f32 layout: VGPR r, M = 16*mt + r + 8*grp, N = ncol
  int mbase = mt * 16 + grp * 8;
#pragma unroll
  for (int r = 0; r < 8; ++r)
    C[(size_t)(mbase + r) * Ncol + ncol] = acc[r];
}

// ---------------- per-node attention coefficients ----------------
// as[n,h] = dot(H[n, h*ch:(h+1)*ch], a_src[h]); same for ad.
__global__ void alpha_kernel(const float* __restrict__ H, const float* __restrict__ a_src,
                             const float* __restrict__ a_dst, float* __restrict__ as,
                             float* __restrict__ ad, int heads, int ch) {
  int idx = blockIdx.x * blockDim.x + threadIdx.x;
  if (idx >= NN * heads) return;
  int node = idx / heads, h = idx - node * heads;
  const float* row = H + (size_t)node * heads * ch + (size_t)h * ch;
  const float* vs = a_src + (size_t)h * ch;
  const float* vd = a_dst + (size_t)h * ch;
  float ss = 0.0f, sd = 0.0f;
  for (int c = 0; c < ch; ++c) { float v = row[c]; ss += v * vs[c]; sd += v * vd[c]; }
  as[idx] = ss;
  ad[idx] = sd;
}

// ---------------- segment softmax over incoming edges ----------------
__global__ void edge_max_kernel(const int* __restrict__ ei, const float* __restrict__ as,
                                const float* __restrict__ ad, float* __restrict__ m, int heads) {
  int idx = blockIdx.x * blockDim.x + threadIdx.x;
  if (idx >= ETOT * heads) return;
  int e = idx / heads, h = idx - e * heads;
  int s, d; edge_sd(ei, e, s, d);
  float v = as[s * heads + h] + ad[d * heads + h];
  v = v > 0.0f ? v : v * NEG_SLOPE;
  atomic_max_f32(&m[d * heads + h], v);
}

__global__ void edge_exp_kernel(const int* __restrict__ ei, const float* __restrict__ as,
                                const float* __restrict__ ad, const float* __restrict__ m,
                                float* __restrict__ ex, float* __restrict__ denom, int heads) {
  int idx = blockIdx.x * blockDim.x + threadIdx.x;
  if (idx >= ETOT * heads) return;
  int e = idx / heads, h = idx - e * heads;
  int s, d; edge_sd(ei, e, s, d);
  float v = as[s * heads + h] + ad[d * heads + h];
  v = v > 0.0f ? v : v * NEG_SLOPE;
  float mm = m[d * heads + h];
  if (!isfinite(mm)) mm = 0.0f;                 // mirrors jnp.where(isfinite(m), m, 0)
  float x = expf(v - mm);
  ex[idx] = x;
  atomicAdd(&denom[d * heads + h], x);
}

// out[dst, f:f+4] += H[src, f:f+4] * ex[e,h]/(denom[dst,h]+eps)
// One thread per (edge, 4-feature chunk): b128 gather + 4 native f32 atomics.
// FDIM==128 for both layers -> 32 chunks/edge; a 4-aligned chunk never
// straddles a head boundary (ch = 32 or 128).
__global__ void edge_agg_kernel(const int* __restrict__ ei, const float* __restrict__ H,
                                const float* __restrict__ ex, const float* __restrict__ denom,
                                float* __restrict__ out, int heads, int ch) {
  int idx = blockIdx.x * blockDim.x + threadIdx.x;
  if (idx >= ETOT * (FDIM / 4)) return;
  int e = idx >> 5;                 // FDIM/4 == 32 chunks per edge
  int f = (idx & 31) << 2;
  int h = f / ch;
  int s, d; edge_sd(ei, e, s, d);
  float a = ex[e * heads + h] / (denom[d * heads + h] + SM_EPS);
  const float4 hv = *(const float4*)(H + (size_t)s * FDIM + f);
  float* o = out + (size_t)d * FDIM + f;
  atomicAdd(o + 0, hv.x * a);
  atomicAdd(o + 1, hv.y * a);
  atomicAdd(o + 2, hv.z * a);
  atomicAdd(o + 3, hv.w * a);
}

// ---------------- elementwise tails ----------------
// buf -> elu(buf + b1), emitted directly as the layer-2 GEMM's hi/lo f16 split
// (the post-ELU activations are consumed only by the layer-2 GEMM).
__global__ void bias_elu_split_kernel(const float* __restrict__ buf, const float* __restrict__ bias,
                                      _Float16* __restrict__ hi, _Float16* __restrict__ lo, int n) {
  int i = blockIdx.x * blockDim.x + threadIdx.x;
  if (i >= n) return;
  float v = buf[i] + bias[i & (FDIM - 1)];
  v = v > 0.0f ? v : expm1f(v);                 // ELU
  _Float16 h = (_Float16)v;
  hi[i] = h;
  lo[i] = (_Float16)(v - (float)h);
}

__global__ void bias_add_kernel(float* __restrict__ buf, const float* __restrict__ bias, int n) {
  int i = blockIdx.x * blockDim.x + threadIdx.x;
  if (i >= n) return;
  buf[i] += bias[i & (FDIM - 1)];
}

// ---------------- host orchestration ----------------
extern "C" void kernel_launch(void* const* d_in, const int* in_sizes, int n_in,
                              void* d_out, int out_size, void* d_ws, size_t ws_size,
                              hipStream_t stream) {
  const float* x   = (const float*)d_in[0];
  const int*   ei  = (const int*)  d_in[1];   // [2, EE] flat: [0..EE)=src, [EE..2EE)=dst
  const float* W1  = (const float*)d_in[2];
  const float* as1 = (const float*)d_in[3];
  const float* ad1 = (const float*)d_in[4];
  const float* b1  = (const float*)d_in[5];
  const float* W2  = (const float*)d_in[6];
  const float* as2 = (const float*)d_in[7];
  const float* ad2 = (const float*)d_in[8];
  const float* b2  = (const float*)d_in[9];
  float* out = (float*)d_out;

  // workspace carve-up (~57 MB; resident in MI455X's 192 MB L2)
  char* ws = (char*)d_ws;
  size_t off = 0;
  auto take = [&](size_t bytes) -> char* {
    char* p = ws + off;
    off = (off + bytes + 255) & ~(size_t)255;
    return p;
  };
  _Float16* Ahi = (_Float16*)take((size_t)NN * FDIM * 2);
  _Float16* Alo = (_Float16*)take((size_t)NN * FDIM * 2);
  _Float16* Bhi = (_Float16*)take((size_t)FDIM * FDIM * 2);
  _Float16* Blo = (_Float16*)take((size_t)FDIM * FDIM * 2);
  float* featA = (float*)take((size_t)NN * FDIM * 4);   // GEMM output (h)
  float* featB = (float*)take((size_t)NN * FDIM * 4);   // layer-1 aggregation accumulator
  float* asv   = (float*)take((size_t)NN * 4 * 4);
  float* adv   = (float*)take((size_t)NN * 4 * 4);
  float* mv    = (float*)take((size_t)NN * 4 * 4);
  float* dv    = (float*)take((size_t)NN * 4 * 4);
  float* exv   = (float*)take((size_t)ETOT * 4 * 4);
  (void)in_sizes; (void)n_in; (void)out_size; (void)ws_size;

  const int B = 256;                                    // 8 wave32 per block
  auto g = [](long long n, int b) { return (unsigned)((n + b - 1) / b); };
  const unsigned gemm_blocks = (NN / 16) * (FDIM / 16) / 8;   // 1 wave per 16x16 tile

  // ================= layer 1 (4 heads x 32) =================
  split_kernel <<<g((long long)NN * FDIM, B), B, 0, stream>>>(x,  Ahi, Alo, NN * FDIM);
  splitT_kernel<<<g(FDIM * FDIM, B),          B, 0, stream>>>(W1, Bhi, Blo, FDIM, FDIM);
  wmma_gemm_kernel<<<gemm_blocks, B, 0, stream>>>(Ahi, Alo, Bhi, Blo, featA, NN, FDIM, FDIM);
  alpha_kernel<<<g(NN * 4, B), B, 0, stream>>>(featA, as1, ad1, asv, adv, 4, 32);
  fill_kernel <<<g(NN * 4, B), B, 0, stream>>>(mv, -INFINITY, NN * 4);
  fill_kernel <<<g(NN * 4, B), B, 0, stream>>>(dv, 0.0f, NN * 4);
  fill_kernel <<<g((long long)NN * FDIM, B), B, 0, stream>>>(featB, 0.0f, NN * FDIM);
  edge_max_kernel<<<g((long long)ETOT * 4, B), B, 0, stream>>>(ei, asv, adv, mv, 4);
  edge_exp_kernel<<<g((long long)ETOT * 4, B), B, 0, stream>>>(ei, asv, adv, mv, exv, dv, 4);
  edge_agg_kernel<<<g((long long)ETOT * (FDIM / 4), B), B, 0, stream>>>(ei, featA, exv, dv, featB, 4, 32);
  // ELU + bias fused with the layer-2 precision split
  bias_elu_split_kernel<<<g((long long)NN * FDIM, B), B, 0, stream>>>(featB, b1, Ahi, Alo, NN * FDIM);

  // ================= layer 2 (1 head x 128, mean==identity) =================
  splitT_kernel<<<g(FDIM * FDIM, B), B, 0, stream>>>(W2, Bhi, Blo, FDIM, FDIM);
  wmma_gemm_kernel<<<gemm_blocks, B, 0, stream>>>(Ahi, Alo, Bhi, Blo, featA, NN, FDIM, FDIM);
  alpha_kernel<<<g(NN, B), B, 0, stream>>>(featA, as2, ad2, asv, adv, 1, FDIM);
  fill_kernel <<<g(NN, B), B, 0, stream>>>(mv, -INFINITY, NN);
  fill_kernel <<<g(NN, B), B, 0, stream>>>(dv, 0.0f, NN);
  fill_kernel <<<g((long long)NN * FDIM, B), B, 0, stream>>>(out, 0.0f, NN * FDIM);
  edge_max_kernel<<<g((long long)ETOT, B), B, 0, stream>>>(ei, asv, adv, mv, 1);
  edge_exp_kernel<<<g((long long)ETOT, B), B, 0, stream>>>(ei, asv, adv, mv, exv, dv, 1);
  edge_agg_kernel<<<g((long long)ETOT * (FDIM / 4), B), B, 0, stream>>>(ei, featA, exv, dv, out, 1, FDIM);
  bias_add_kernel<<<g((long long)NN * FDIM, B), B, 0, stream>>>(out, b2, NN * FDIM);
}
